// PointLSTMBlock_5411658793009
// MI455X (gfx1250) — compile-verified, matrix operand order checked by reading, not run on hardware
//
#include <hip/hip_runtime.h>
#include <hip/hip_bf16.h>

// ---------------------------------------------------------------------------
// PointLSTM block for MI455X (gfx1250, wave32, WMMA).
//
// B=4, T=32, N=1024, K=16, HD=128.  Gates GEMM (512 x {136|264} x N*K) is
// ~860 GFLOP total -> f16 WMMA (V_WMMA_F32_16X16X32_F16), K padded 160/288.
// Wave w owns M-tiles {w, w+8, w+16, w+24} = i/f/o/g rows of channels
// [16w,16w+16): cell update is per-lane in registers, max-over-K is a
// shfl_xor butterfly.  4 points per WG amortize L2 weight streaming 4x.
// __launch_bounds__(256,1): 2 waves/SIMD -> full VGPR budget, no acc spills.
// ---------------------------------------------------------------------------

#define BB   4
#define TT   32
#define NN   1024
#define KK   16
#define HD_  128
#define NPW  4      // points per workgroup

typedef __attribute__((ext_vector_type(16))) _Float16 v16h;
typedef __attribute__((ext_vector_type(8)))  _Float16 v8h;
typedef __attribute__((ext_vector_type(8)))  float    v8f;

__device__ __forceinline__ int XI(int b, int t, int c, int n) {
    return ((b * TT + t) * 4 + c) * NN + n;   // x layout (B,T,4,N)
}
__device__ __forceinline__ float sigm(float v) {
    return 1.f / (1.f + __expf(-v));
}
__device__ __forceinline__ float tanh_fast(float v) {
    float a = fabsf(v);
    float e = __expf(-2.f * a);
    float r = (1.f - e) / (1.f + e);
    return copysignf(r, v);
}

// ---------------------------------------------------------------------------
// KNN: one workgroup per (b,t); refs in LDS; 8 waves, one query each per
// round; 16 selection rounds with wave-wide shfl_xor min-reduce.
// ---------------------------------------------------------------------------
__global__ void knn_kernel(const float* __restrict__ x, int* __restrict__ gind) {
    __shared__ float refs[3 * NN];
    __shared__ float d2buf[8 * NN];
    const int b = blockIdx.y, t = blockIdx.x;
    const int tr = (t == 0) ? 0 : (t - 1);
    const int tid = threadIdx.x;

    for (int i = tid; i < NN; i += 256) {
        refs[i]          = x[XI(b, tr, 0, i)];
        refs[NN + i]     = x[XI(b, tr, 1, i)];
        refs[2 * NN + i] = x[XI(b, tr, 2, i)];
    }
    __syncthreads();

    const int lane = tid & 31, wv = tid >> 5;
    for (int q = wv; q < NN; q += 8) {
        const float qx = x[XI(b, t, 0, q)];
        const float qy = x[XI(b, t, 1, q)];
        const float qz = x[XI(b, t, 2, q)];
        for (int c = lane; c < NN; c += 32) {
            float dx = qx - refs[c];
            float dy = qy - refs[NN + c];
            float dz = qz - refs[2 * NN + c];
            d2buf[wv * NN + c] = dx * dx + dy * dy + dz * dz;
        }
        asm volatile("s_wait_dscnt 0" ::: "memory");
        for (int sel = 0; sel < KK; ++sel) {
            float best = 3.0e38f; int bidx = 0x7fffffff;
            for (int c = lane; c < NN; c += 32) {
                float v = d2buf[wv * NN + c];
                if (v < best) { best = v; bidx = c; }
            }
            for (int off = 16; off; off >>= 1) {
                float ov = __shfl_xor(best, off, 32);
                int   oi = __shfl_xor(bidx, off, 32);
                if (ov < best || (ov == best && oi < bidx)) { best = ov; bidx = oi; }
            }
            if (lane == 0) {
                gind[((b * TT + t) * NN + q) * KK + sel] = bidx;
                d2buf[wv * NN + bidx] = 3.0e38f;
            }
            asm volatile("s_wait_dscnt 0" ::: "memory");
        }
    }
}

// Weight fp32 -> f16, K-padded with zeros (row-major 512 x KDP).
__global__ void convert_w(const float* __restrict__ src, _Float16* __restrict__ dst,
                          int KD, int KDP) {
    int i = blockIdx.x * 256 + threadIdx.x;
    if (i >= 512 * KDP) return;
    int r = i / KDP, kk = i - r * KDP;
    dst[i] = (kk < KD) ? (_Float16)src[r * KD + kk] : (_Float16)0.f;
}

// out1 position channels: out1[bt,0:4,n] = x[bt,0:4,n]
__global__ void pos_copy(const float* __restrict__ x, float* __restrict__ out1) {
    int i = blockIdx.x * 256 + threadIdx.x;
    if (i >= BB * TT * 4 * NN) return;
    int n = i & (NN - 1);
    int rest = i >> 10;
    int c = rest & 3;
    int bt = rest >> 2;
    out1[(bt * 132 + c) * NN + n] = x[i];
}

// ---------------------------------------------------------------------------
// One LSTM time step.  Grid (N/4, B), 256 threads (8 wave32).
// WG handles 4 points: comb (KDP x 64, f16 col-major) + cg (f16) in LDS,
// 16 WMMA per K-chunk (4 gate M-tiles x 4 point N-tiles), in-register
// epilogue with shfl_xor max-over-16-neighbors.
// ---------------------------------------------------------------------------
template <int LAYER>
__global__ __launch_bounds__(256, 1)
void lstm_step(const float* __restrict__ x,
               const _Float16* __restrict__ wh,
               const float* __restrict__ bias,
               const int* __restrict__ gind,
               const float* __restrict__ hs0_in,
               const float* __restrict__ h_prev,
               const float* __restrict__ c_prev,
               float* __restrict__ h_next,
               float* __restrict__ c_next,
               float* __restrict__ hs_out,
               int t) {
    constexpr int KDP    = (LAYER == 0) ? 160 : 288;   // padded K
    constexpr int NCH    = KDP / 32;                   // WMMA K-chunks
    constexpr int CSTR   = KDP + 16;                   // comb col stride (halves)
    constexpr int CG_STR = 136;                        // cg stride (halves, pad 8)
    constexpr int NCOL   = NPW * KK;                   // 64 columns

    __shared__ __align__(16) _Float16 combT[NCOL * CSTR]; // B-matrix, col-major
    __shared__ __align__(16) _Float16 cgH[NCOL * CG_STR]; // gathered c (f16)
    __shared__ float biasS[512];
    __shared__ int   sI[NCOL];

    const int b   = blockIdx.y;
    const int n0  = blockIdx.x * NPW;
    const int tid = threadIdx.x;
    const int tp  = (t > 0) ? (t - 1) : 0;

    __builtin_prefetch(wh, 0, 3);   // global_prefetch: weights live in L2

    for (int i = tid; i < 512; i += 256) biasS[i] = bias[i];
    if (tid < NCOL)
        sI[tid] = gind[((b * TT + t) * NN + (n0 + (tid >> 4))) * KK + (tid & 15)];
    __syncthreads();

    // ---- build comb (f16) and cg (f16) in LDS -----------------------------
    {
        const int col = tid >> 2;      // 0..63
        const int sub = tid & 3;
        const int p   = col >> 4;
        const int n   = n0 + p;
        const int gi  = sI[col];
        for (int r = sub; r < KDP; r += 4) {
            float v;
            if constexpr (LAYER == 0) {
                if (r < 4)        v = x[XI(b, t, r, n)];
                else if (r < 8)   v = x[XI(b, tp, r - 4, gi)] - x[XI(b, t, r - 4, n)];
                else if (r < 136) v = (t > 0) ? h_prev[(b * HD_ + (r - 8)) * NN + gi] : 0.f;
                else              v = 0.f;
            } else {
                if (r < 4)        v = x[XI(b, t, r, n)];
                else if (r < 132) v = hs0_in[((b * TT + t) * HD_ + (r - 4)) * NN + n];
                else if (r < 136) v = x[XI(b, tp, r - 132, gi)] - x[XI(b, t, r - 132, n)];
                else if (r < 264) v = (t > 0) ? h_prev[(b * HD_ + (r - 136)) * NN + gi] : 0.f;
                else              v = 0.f;
            }
            combT[col * CSTR + r] = (_Float16)v;
        }
        for (int i2 = tid; i2 < NCOL * HD_; i2 += 256) {
            int cc = i2 >> 7, ch = i2 & 127;
            float cv = (t > 0) ? c_prev[(b * HD_ + ch) * NN + sI[cc]] : 0.f;
            cgH[cc * CG_STR + ch] = (_Float16)cv;
        }
    }
    __syncthreads();

    // ---- gates GEMM: wave w owns gate tiles {w,w+8,w+16,w+24}, 4 N-tiles --
    const int lane = tid & 31;
    const int wv   = tid >> 5;
    const int lrow = lane & 15;
    const int kofs = (lane >> 4) << 3;   // ISA 16-bit A/B layout: hi lanes +8

    const v8f vzero = {0.f, 0.f, 0.f, 0.f, 0.f, 0.f, 0.f, 0.f};
    v8f acc[4][NPW];                     // [gate][point]
#pragma unroll
    for (int g = 0; g < 4; ++g)
#pragma unroll
        for (int p = 0; p < NPW; ++p) acc[g][p] = vzero;

    // one base pointer per lane; gate rows become immediate offsets
    const _Float16* wbase = wh + (wv * 16 + lrow) * KDP;

#pragma unroll
    for (int kt = 0; kt < NCH; ++kt) {
        const int kb = kt * 32 + kofs;
        // keep only the 4 B fragments resident; A fragments are transient
        v16h bf[NPW];
#pragma unroll
        for (int p = 0; p < NPW; ++p) {
            const _Float16* bp = &combT[(p * 16 + lrow) * CSTR + kb];
            v8h blo = *(const v8h*)bp;
            v8h bhi = *(const v8h*)(bp + 16);
            bf[p] = __builtin_shufflevector(blo, bhi,
                0,1,2,3,4,5,6,7,8,9,10,11,12,13,14,15);
        }
#pragma unroll
        for (int g = 0; g < 4; ++g) {
            const _Float16* ap = wbase + (g * 128) * KDP + kb;
            v8h alo = *(const v8h*)ap;
            v8h ahi = *(const v8h*)(ap + 16);
            v16h af = __builtin_shufflevector(alo, ahi,
                0,1,2,3,4,5,6,7,8,9,10,11,12,13,14,15);
#pragma unroll
            for (int p = 0; p < NPW; ++p)
                acc[g][p] = __builtin_amdgcn_wmma_f32_16x16x32_f16(
                    false, af, false, bf[p], (short)0, acc[g][p], false, false);
        }
    }

    // ---- in-register cell update + shfl max over the 16 neighbors ---------
    // C/D layout: vgpr r, lanes 0-15 -> M=r, N=lane; lanes 16-31 -> M=r+8.
    const int k  = lane & 15;            // neighbor index
    const int hi = lane >> 4;
#pragma unroll
    for (int p = 0; p < NPW; ++p) {
        const int n = n0 + p;
#pragma unroll
        for (int r = 0; r < 8; ++r) {
            const int ch = wv * 16 + hi * 8 + r;
            float ii = acc[0][p][r] + biasS[ch];
            float ff = acc[1][p][r] + biasS[ch + 128];
            float oo = acc[2][p][r] + biasS[ch + 256];
            float gg = acc[3][p][r] + biasS[ch + 384];
            float cg = (float)cgH[(p * 16 + k) * CG_STR + ch];
            float cn = sigm(ff) * cg + sigm(ii) * tanh_fast(gg);
            float hn = sigm(oo) * tanh_fast(cn);
#pragma unroll
            for (int off = 1; off < 16; off <<= 1) {
                hn = fmaxf(hn, __shfl_xor(hn, off, 32));
                cn = fmaxf(cn, __shfl_xor(cn, off, 32));
            }
            if (k == 0) {                // lanes 0 and 16 write their channel
                h_next[(b * HD_ + ch) * NN + n] = hn;
                c_next[(b * HD_ + ch) * NN + n] = cn;
                if constexpr (LAYER == 0)
                    hs_out[((b * TT + t) * HD_ + ch) * NN + n] = hn;       // hs0
                else
                    hs_out[((b * TT + t) * 132 + 4 + ch) * NN + n] = hn;   // out1
            }
        }
    }
}

// ---------------------------------------------------------------------------
// Host-side launch.  d_out = [out1 | h | c | group_ind] flat.
// ws layout (bytes): w0h | w1h | hs0 | hbuf[2] | cbuf[2]  (~72.5 MiB).
// ---------------------------------------------------------------------------
extern "C" void kernel_launch(void* const* d_in, const int* in_sizes, int n_in,
                              void* d_out, int out_size, void* d_ws, size_t ws_size,
                              hipStream_t stream) {
    (void)in_sizes; (void)n_in; (void)out_size; (void)ws_size;

    const float* x  = (const float*)d_in[0];
    const float* w0 = (const float*)d_in[1];
    const float* b0 = (const float*)d_in[2];
    const float* w1 = (const float*)d_in[3];
    const float* b1 = (const float*)d_in[4];

    constexpr size_t OUT1_E = (size_t)BB * TT * 132 * NN;   // 17,301,504
    constexpr size_t H_E    = (size_t)BB * HD_ * NN;        //    524,288

    float* out1 = (float*)d_out;
    float* hout = out1 + OUT1_E;
    float* cout = hout + H_E;
    int*   gind = (int*)(cout + H_E);

    char* ws = (char*)d_ws;
    _Float16* w0h = (_Float16*)ws;                        // 512*160*2 = 163,840
    _Float16* w1h = (_Float16*)(ws + 163840);             // 512*288*2 = 294,912
    float* hs0    = (float*)(ws + 458752);                // B*T*HD*N*4 = 64 MiB
    float* hb0    = (float*)(ws + 67567616);
    float* hb1    = (float*)(ws + 69664768);
    float* cb0    = (float*)(ws + 71761920);
    float* cb1    = (float*)(ws + 73859072);
    float* hb[2] = {hb0, hb1};
    float* cb[2] = {cb0, cb1};

    knn_kernel<<<dim3(TT, BB), 256, 0, stream>>>(x, gind);
    convert_w<<<(512 * 160 + 255) / 256, 256, 0, stream>>>(w0, w0h, 136, 160);
    convert_w<<<(512 * 288 + 255) / 256, 256, 0, stream>>>(w1, w1h, 264, 288);
    pos_copy<<<(BB * TT * 4 * NN + 255) / 256, 256, 0, stream>>>(x, out1);

    for (int t = 0; t < TT; ++t)
        lstm_step<0><<<dim3(NN / NPW, BB), 256, 0, stream>>>(
            x, w0h, b0, gind, nullptr,
            hb[t & 1], cb[t & 1], hb[(t + 1) & 1], cb[(t + 1) & 1], hs0, t);

    for (int t = 0; t < TT; ++t)
        lstm_step<1><<<dim3(NN / NPW, BB), 256, 0, stream>>>(
            x, w1h, b1, gind, hs0,
            hb[t & 1], cb[t & 1], hb[(t + 1) & 1], cb[(t + 1) & 1], out1, t);

    // T=32 (even): final state lands in buffer 0.
    hipMemcpyAsync(hout, hb[0], H_E * sizeof(float), hipMemcpyDeviceToDevice, stream);
    hipMemcpyAsync(cout, cb[0], H_E * sizeof(float), hipMemcpyDeviceToDevice, stream);
}
